// MultiHeadAttention_32547262169120
// MI455X (gfx1250) — matrix operand order repaired and checked by reading, not compiled, not run on hardware
//
#include <hip/hip_runtime.h>

typedef __attribute__((ext_vector_type(16))) _Float16 v16h;
typedef __attribute__((ext_vector_type(8)))  _Float16 v8h;
typedef __attribute__((ext_vector_type(8)))  float    v8f;
typedef __attribute__((ext_vector_type(4)))  unsigned int v4u;
typedef __attribute__((ext_vector_type(8)))  int      v8i;
typedef __attribute__((ext_vector_type(4)))  int      v4i;

#define EMBED 2048
#define HEADS 16
#define HDIM  128
#define BATCH 2
#define SEQ   2048
#define MTOT  (BATCH * SEQ)   // 4096 tokens

#define TM 128
#define TN 128
#define TK 32
#define LDS_LD (TK + 8)       // padded leading dim (f16 elems) = 40; 64B row + 16B pad

// ---------------------------------------------------------------------------
// WMMA wrapper: D = A(16x32 f16) * B(32x16 f16) + C(16x16 f32)
// ---------------------------------------------------------------------------
__device__ __forceinline__ v8f wmma_f16(v16h a, v16h b, v8f c) {
  return __builtin_amdgcn_wmma_f32_16x16x32_f16(
      /*neg_a=*/false, a, /*neg_b=*/false, b,
      /*c_mod=*/(short)0, c, /*reuse_a=*/false, /*reuse_b=*/false);
}

// A-fragment: 16x32 f16 tile, row-major, leading dim `ld` elements.
// Contiguous k order used here (same in B loaders): lane<16 holds row m=lane, K{0..15}; lane>=16 K{16..31}.
__device__ __forceinline__ v16h load_a_frag(const _Float16* p, int ld) {
  const int lane = threadIdx.x & 31;
  const _Float16* r = p + (lane & 15) * ld + ((lane < 16) ? 0 : 16);
  v16h a;
#pragma unroll
  for (int i = 0; i < 4; ++i) {
    a[2 * i]     = r[2 * i];
    a[2 * i + 1] = r[2 * i + 1];
    a[8 + 2 * i]     = r[8 + 2 * i];
    a[8 + 2 * i + 1] = r[8 + 2 * i + 1];
  }
  return a;
}

// B-fragment from memory holding B^T (rows = N(16), cols = Kdim(32), ld elems).
__device__ __forceinline__ v16h load_bt_frag(const _Float16* p, int ld) {
  const int lane = threadIdx.x & 31;
  const _Float16* r = p + (lane & 15) * ld + ((lane < 16) ? 0 : 16);
  v16h b;
#pragma unroll
  for (int i = 0; i < 16; ++i) b[i] = r[i];
  return b;
}

// B-fragment from row-major B (rows = Kdim(32), cols = N, ld elems).
__device__ __forceinline__ v16h load_b_frag(const _Float16* p, int ld) {
  const int lane = threadIdx.x & 31;
  const int n  = lane & 15;
  const int kb = (lane < 16) ? 0 : 16;
  v16h b;
#pragma unroll
  for (int i = 0; i < 16; ++i) b[i] = p[(kb + i) * ld + n];
  return b;
}

// ---------------------------------------------------------------------------
// TDM: 2D tile load Global -> LDS with per-row LDS padding (ISA ch.8 D#).
// f16 elements (data_size=1 -> 2 bytes). One issue per wave; EXEC ignored.
// clang-23 toolchain: 6-arg builtin (g0, g1, g2, g3, g4, cpol).
// ---------------------------------------------------------------------------
__device__ __forceinline__ void tdm_load_2d_f16(
    unsigned lds_byte_addr, const void* gaddr,
    unsigned tile_d0, unsigned tile_d1,
    unsigned tensor_d0, unsigned tensor_d1,
    unsigned long long stride0_elems,
    unsigned pad_interval, unsigned pad_amount, unsigned pad_en)
{
  const unsigned long long ga = (unsigned long long)(uintptr_t)gaddr;
  v4u g0;
  g0[0] = 1u;                                             // count=1 (valid), user-mode load
  g0[1] = lds_byte_addr;                                  // [63:32] lds_addr
  g0[2] = (unsigned)(ga & 0xFFFFFFFFu);                   // [95:64] global_addr lo
  g0[3] = (unsigned)((ga >> 32) & 0x01FFFFFFu)            // [120:96] global_addr hi
        | (2u << 30);                                     // [127:126] type=2 (image)
  v8i g1;
  g1[0] = (int)((1u << 16)                                // data_size=1 -> 2 bytes
        | (pad_en << 20)                                  // pad_enable
        | (pad_interval << 22)                            // 3 -> every 16 DWORDs (64B row)
        | (pad_amount << 25));                            // 3 -> 4 DWORDs (16B) pad
  g1[1] = (int)((tensor_d0 & 0xFFFFu) << 16);             // [63:48] tensor_dim0 lo16
  g1[2] = (int)(((tensor_d0 >> 16) & 0xFFFFu)             // [79:64] tensor_dim0 hi16
        | ((tensor_d1 & 0xFFFFu) << 16));                 // [95:80] tensor_dim1 lo16
  g1[3] = (int)(((tensor_d1 >> 16) & 0xFFFFu)             // [111:96] tensor_dim1 hi16
        | ((tile_d0 & 0xFFFFu) << 16));                   // [127:112] tile_dim0
  g1[4] = (int)(tile_d1 & 0xFFFFu);                       // [143:128] tile_dim1; tile_dim2=0
  g1[5] = (int)(stride0_elems & 0xFFFFFFFFu);             // [191:160] dim0 stride lo
  g1[6] = (int)((stride0_elems >> 32) & 0xFFFFu);         // [207:192] stride hi; dim1_stride=0
  g1[7] = 0;
  const v4i z4 = {0, 0, 0, 0};                            // 2D tensor: groups 2/3 null
  const v8i z8 = {0, 0, 0, 0, 0, 0, 0, 0};
  __builtin_amdgcn_tensor_load_to_lds(g0, g1, z4, z4, z8, 0);
}

// ---------------------------------------------------------------------------
// fp32 -> f16 convert
// ---------------------------------------------------------------------------
__global__ __launch_bounds__(256) void cvt_f32_to_f16(
    const float* __restrict__ in, _Float16* __restrict__ out, int nElem) {
  int i = blockIdx.x * 256 + threadIdx.x;
  if (i < nElem) out[i] = (_Float16)in[i];
}

// ---------------------------------------------------------------------------
// Tiled f16 GEMM: C[M][N] = A[M][K] * B[K][N] + bias
// A tile staged by the Tensor Data Mover; B tile staged manually (transposed).
// mode 0: write fp32 C.   mode 1: scatter f16 into Q/K/V [B,H,S,D] buffers.
// ---------------------------------------------------------------------------
__global__ __launch_bounds__(256) void mha_gemm_f16(
    const _Float16* __restrict__ A,
    const _Float16* __restrict__ B,
    const float*    __restrict__ bias,
    float*          __restrict__ Cf,
    _Float16* __restrict__ Qb, _Float16* __restrict__ Kb, _Float16* __restrict__ Vb,
    int M, int N, int K, int mode)
{
  __shared__ _Float16 lA[TM][LDS_LD];
  __shared__ _Float16 lB[TN][LDS_LD];   // stored transposed: [n][k]

  const int tid  = threadIdx.x;
  const int wave = tid >> 5;
  const int lane = tid & 31;
  const int bm = blockIdx.y * TM;
  const int bn = blockIdx.x * TN;
  // LDS aperture flat address: low 32 bits are the LDS byte offset
  const unsigned lA_lds = (unsigned)(uintptr_t)(void*)&lA[0][0];

  v8f acc[8];
#pragma unroll
  for (int t = 0; t < 8; ++t)
#pragma unroll
    for (int j = 0; j < 8; ++j) acc[t][j] = 0.0f;

  for (int k0 = 0; k0 < K; k0 += TK) {
    __syncthreads();
    // A tile (128 rows x 32 f16) via TDM; pad 16B per 64B row -> LDS_LD layout.
    if (wave == 0) {
      tdm_load_2d_f16(lA_lds, A + (size_t)bm * K + k0,
                      /*tile*/ TK, TM,
                      /*tensor*/ (unsigned)K, (unsigned)M,
                      /*stride0*/ (unsigned long long)K,
                      /*pad_interval=*/3, /*pad_amount=*/3, /*pad_en=*/1);
      __builtin_amdgcn_s_wait_tensorcnt(0);
    }
    // B tile (32x128) transposed into [n][k] so fragments read contiguous pairs.
    for (int i = tid; i < TK * (TN / 8); i += 256) {
      int kk = i >> 4;
      int nc = (i & 15) * 8;
      v8h vb = *reinterpret_cast<const v8h*>(B + (size_t)(k0 + kk) * N + bn + nc);
#pragma unroll
      for (int j = 0; j < 8; ++j) lB[nc + j][kk] = vb[j];
    }
    __syncthreads();

    if (k0 + TK < K)  // hint next A tile into cache (global_prefetch_b8)
      __builtin_prefetch(A + (size_t)(bm + (tid >> 1)) * K + k0 + TK, 0, 0);

    v16h af = load_a_frag(&lA[wave * 16][0], LDS_LD);
#pragma unroll
    for (int t = 0; t < 8; ++t) {
      v16h bf = load_bt_frag(&lB[t * 16][0], LDS_LD);
      acc[t] = wmma_f16(af, bf, acc[t]);
    }
  }

  const int n  = lane & 15;
  const int rb = (lane < 16) ? 0 : 8;
#pragma unroll
  for (int t = 0; t < 8; ++t) {
    const int col = bn + t * 16 + n;
    const float bv = bias[col];
#pragma unroll
    for (int j = 0; j < 8; ++j) {
      const int row = bm + wave * 16 + rb + j;
      const float v = acc[t][j] + bv;
      if (mode == 0) {
        Cf[(size_t)row * N + col] = v;
      } else {
        const int mat = col >> 11;        // 0=Q 1=K 2=V
        const int rem = col & 2047;
        const int h = rem >> 7;
        const int d = rem & 127;
        _Float16* dst = (mat == 0) ? Qb : (mat == 1) ? Kb : Vb;
        const int bb = row >> 11;         // batch
        const int s  = row & 2047;        // seq pos
        dst[(((size_t)bb * HEADS + h) * SEQ + s) * HDIM + d] = (_Float16)v;
      }
    }
  }
}

// ---------------------------------------------------------------------------
// Flash attention: block = (b,h, 128 q rows); wave owns 16 q rows.
// Loops over 32-key tiles (causal: only (qt+1)*4 tiles visited).
// ---------------------------------------------------------------------------
__global__ __launch_bounds__(256) void mha_attn(
    const _Float16* __restrict__ Q,
    const _Float16* __restrict__ Kc,
    const _Float16* __restrict__ Vc,
    _Float16* __restrict__ ctx)           // [B][S][E], feature = h*128+d
{
  __shared__ _Float16 lP[8][16][LDS_LD];  // per-wave 16x32 P tile (C->A relayout)

  const int tid  = threadIdx.x;
  const int wave = tid >> 5;
  const int lane = tid & 31;
  const int bh = blockIdx.x;              // b*HEADS + h
  const int b  = bh >> 4;
  const int h  = bh & 15;
  const int qt = blockIdx.y;
  const int q0 = qt * 128 + wave * 16;

  const _Float16* Qp = Q  + (size_t)bh * SEQ * HDIM;
  const _Float16* Kp = Kc + (size_t)bh * SEQ * HDIM;
  const _Float16* Vp = Vc + (size_t)bh * SEQ * HDIM;

  // Resident Q fragments: 16 rows x 128 d, as 4 A-frags of K=32.
  v16h qf[4];
#pragma unroll
  for (int i = 0; i < 4; ++i)
    qf[i] = load_a_frag(Qp + (size_t)q0 * HDIM + i * 32, HDIM);

  v8f acc[8];
  float rmax[8], rsum[8];
#pragma unroll
  for (int t = 0; t < 8; ++t)
#pragma unroll
    for (int j = 0; j < 8; ++j) acc[t][j] = 0.0f;
#pragma unroll
  for (int j = 0; j < 8; ++j) { rmax[j] = -1e30f; rsum[j] = 0.0f; }

  const int n  = lane & 15;
  const int rb = (lane < 16) ? 0 : 8;
  const float scale = 0.08838834764831845f;  // 1/sqrt(HEAD_DIM=128)

  const int nkt = (qt + 1) * 4;              // causal: keys <= end of q tile
  for (int kt = 0; kt < nkt; ++kt) {
    const int k0 = kt * 32;

    // scores S = Q * K^T for 32 keys (two 16-wide WMMA column tiles)
    v8f s0, s1;
#pragma unroll
    for (int j = 0; j < 8; ++j) { s0[j] = 0.0f; s1[j] = 0.0f; }
#pragma unroll
    for (int i = 0; i < 4; ++i) {
      v16h kb0 = load_bt_frag(Kp + (size_t)k0 * HDIM + i * 32, HDIM);
      v16h kb1 = load_bt_frag(Kp + (size_t)(k0 + 16) * HDIM + i * 32, HDIM);
      s0 = wmma_f16(qf[i], kb0, s0);
      s1 = wmma_f16(qf[i], kb1, s1);
    }

    // online softmax; row (rb+j) lives across the 16 lanes of this half
#pragma unroll
    for (int j = 0; j < 8; ++j) {
      const int qrow = q0 + rb + j;
      float x0 = (k0 + n      <= qrow) ? s0[j] * scale : -1e30f;
      float x1 = (k0 + 16 + n <= qrow) ? s1[j] * scale : -1e30f;
      float mx = fmaxf(x0, x1);
#pragma unroll
      for (int off = 1; off < 16; off <<= 1)
        mx = fmaxf(mx, __shfl_xor(mx, off, 32));
      const float nm   = fmaxf(rmax[j], mx);
      const float corr = __expf(rmax[j] - nm);
      const float p0 = __expf(x0 - nm);
      const float p1 = __expf(x1 - nm);
      float ps = p0 + p1;
#pragma unroll
      for (int off = 1; off < 16; off <<= 1)
        ps += __shfl_xor(ps, off, 32);
      rsum[j] = rsum[j] * corr + ps;
      rmax[j] = nm;
#pragma unroll
      for (int t = 0; t < 8; ++t) acc[t][j] *= corr;
      lP[wave][rb + j][n]      = (_Float16)p0;
      lP[wave][rb + j][16 + n] = (_Float16)p1;
    }
    __builtin_amdgcn_wave_barrier();   // same-wave LDS is in-order; fence scheduler

    // ctx += P(16x32) * V(32x128)
    v16h pf = load_a_frag(&lP[wave][0][0], LDS_LD);
#pragma unroll
    for (int t = 0; t < 8; ++t) {
      v16h vf = load_b_frag(Vp + (size_t)k0 * HDIM + t * 16, HDIM);
      acc[t] = wmma_f16(pf, vf, acc[t]);
    }
  }

  // normalize and emit f16 ctx in [B,S,H*D] (ready for out-proj GEMM)
#pragma unroll
  for (int j = 0; j < 8; ++j) {
    const float inv = 1.0f / rsum[j];
    const int qrow = q0 + rb + j;
    _Float16* dst = ctx + ((size_t)b * SEQ + qrow) * EMBED + h * HDIM;
#pragma unroll
    for (int t = 0; t < 8; ++t)
      dst[t * 16 + n] = (_Float16)(acc[t][j] * inv);
  }
}

// ---------------------------------------------------------------------------
// Host-side orchestration
// ---------------------------------------------------------------------------
extern "C" void kernel_launch(void* const* d_in, const int* in_sizes, int n_in,
                              void* d_out, int out_size, void* d_ws, size_t ws_size,
                              hipStream_t stream) {
  const float* hidden = (const float*)d_in[0];
  // d_in[1] = attention_mask: exact causal tril -> applied analytically in-kernel
  const float* w_qkv = (const float*)d_in[2];
  const float* b_qkv = (const float*)d_in[3];
  const float* w_out = (const float*)d_in[4];
  const float* b_out = (const float*)d_in[5];
  float* out = (float*)d_out;
  (void)in_sizes; (void)n_in; (void)out_size; (void)ws_size;

  // Workspace carve-out (~117 MB of f16 staging)
  char* p = (char*)d_ws;
  _Float16* hid_h  = (_Float16*)p; p += (size_t)MTOT * EMBED * 2;
  _Float16* wqkv_h = (_Float16*)p; p += (size_t)EMBED * 3 * EMBED * 2;
  _Float16* wout_h = (_Float16*)p; p += (size_t)EMBED * EMBED * 2;
  _Float16* q_h    = (_Float16*)p; p += (size_t)MTOT * EMBED * 2;
  _Float16* k_h    = (_Float16*)p; p += (size_t)MTOT * EMBED * 2;
  _Float16* v_h    = (_Float16*)p; p += (size_t)MTOT * EMBED * 2;
  _Float16* ctx_h  = (_Float16*)p; p += (size_t)MTOT * EMBED * 2;

  {
    int n = MTOT * EMBED;
    cvt_f32_to_f16<<<(n + 255) / 256, 256, 0, stream>>>(hidden, hid_h, n);
  }
  {
    int n = EMBED * 3 * EMBED;
    cvt_f32_to_f16<<<(n + 255) / 256, 256, 0, stream>>>(w_qkv, wqkv_h, n);
  }
  {
    int n = EMBED * EMBED;
    cvt_f32_to_f16<<<(n + 255) / 256, 256, 0, stream>>>(w_out, wout_h, n);
  }

  // QKV projection, fused bias + per-head scatter
  dim3 g1(3 * EMBED / TN, MTOT / TM);
  mha_gemm_f16<<<g1, 256, 0, stream>>>(hid_h, wqkv_h, b_qkv, nullptr,
                                       q_h, k_h, v_h, MTOT, 3 * EMBED, EMBED, 1);

  // Flash attention
  dim3 ga(BATCH * HEADS, SEQ / 128);
  mha_attn<<<ga, 256, 0, stream>>>(q_h, k_h, v_h, ctx_h);

  // Output projection (fp32 out + bias)
  dim3 g2(EMBED / TN, MTOT / TM);
  mha_gemm_f16<<<g2, 256, 0, stream>>>(ctx_h, wout_h, b_out, out,
                                       nullptr, nullptr, nullptr, MTOT, EMBED, EMBED, 0);
}
